// FreqCondInjection_37555194036421
// MI455X (gfx1250) — matrix operand order repaired and checked by reading, not compiled, not run on hardware
//
#include <hip/hip_runtime.h>
#include <hip/hip_bf16.h>
#include <math.h>

// ---------------------------------------------------------------------------
// FreqCondInjection for MI455X (gfx1250).
//  - GEMM-shaped work on v_wmma_f32_16x16x32_f16 (convs as implicit GEMM,
//    frequency attention as wave32 flash kernel with head_dim == WMMA K).
//  - Conv weight tiles staged into LDS by the Tensor Data Mover
//    (tensor_load_to_lds + s_wait_tensorcnt) when the builtin is available,
//    overlapping the DMA with the activation f32->f16 staging.
//  - All WMMA fragments are wide aligned LDS/global vector loads.
// ---------------------------------------------------------------------------

typedef __attribute__((ext_vector_type(16))) _Float16 v16h;
typedef __attribute__((ext_vector_type(8)))  _Float16 v8h;
typedef __attribute__((ext_vector_type(8)))  float    v8f;
typedef __attribute__((ext_vector_type(4)))  unsigned int u32x4;
typedef __attribute__((ext_vector_type(8)))  int      i32x8;
typedef __attribute__((ext_vector_type(4)))  int      i32x4;

#if __has_builtin(__builtin_amdgcn_tensor_load_to_lds) && \
    __has_builtin(__builtin_amdgcn_s_wait_tensorcnt)
#define USE_TDM 1
#else
#define USE_TDM 0
#endif

#define HH 48
#define WW 48
#define HWP 2304            // 48*48
#define BB 4
#define CC 256
#define SPX 2359296         // B*256*H*W
#define FREQW 25            // rfft width
#define MTOK 1200           // 48*25 frequency tokens
#define FREQN 1228800       // B*256*1200
#define PI2 6.283185307179586f

// ------------------------------ GroupNorm ---------------------------------
__global__ void k_groupnorm(const float* __restrict__ x, const float* __restrict__ gw,
                            const float* __restrict__ gb, float* __restrict__ xn) {
  int bg = blockIdx.x;            // 128 = 4 batches * 32 groups
  int b = bg >> 5, g = bg & 31;
  const size_t base = ((size_t)b * CC + g * 8) * HWP;
  const int n = 8 * HWP;
  float s = 0.f, s2 = 0.f;
  for (int i = threadIdx.x; i < n; i += 256) {
    float v = x[base + i];
    s += v; s2 += v * v;
  }
  __shared__ float sh[256], sh2[256];
  sh[threadIdx.x] = s; sh2[threadIdx.x] = s2;
  __syncthreads();
  for (int o = 128; o > 0; o >>= 1) {
    if (threadIdx.x < o) {
      sh[threadIdx.x]  += sh[threadIdx.x + o];
      sh2[threadIdx.x] += sh2[threadIdx.x + o];
    }
    __syncthreads();
  }
  float mean = sh[0] * (1.f / 18432.f);
  float var  = sh2[0] * (1.f / 18432.f) - mean * mean;
  float rstd = rsqrtf(var + 1e-5f);
  for (int i = threadIdx.x; i < n; i += 256) {
    int c = g * 8 + i / HWP;
    xn[base + i] = (x[base + i] - mean) * rstd * gw[c] + gb[c];
  }
}

// --------------------------- Depthwise 3x3 --------------------------------
__global__ void k_dw3x3(const float* __restrict__ in, const float* __restrict__ w,
                        float* __restrict__ out) {
  size_t i = (size_t)blockIdx.x * 256 + threadIdx.x;
  if (i >= (size_t)SPX) return;
  int xcol = (int)(i % WW);
  int yrow = (int)((i / WW) % HH);
  int c = (int)((i / HWP) % CC);
  const float* wp = w + c * 9;
  float acc = 0.f;
#pragma unroll
  for (int dy = -1; dy <= 1; ++dy) {
#pragma unroll
    for (int dx = -1; dx <= 1; ++dx) {
      int yy = yrow + dy, xx = xcol + dx;
      if (yy >= 0 && yy < HH && xx >= 0 && xx < WW)
        acc += in[i + dy * WW + dx] * wp[(dy + 1) * 3 + (dx + 1)];
    }
  }
  out[i] = acc;
}

// --------------------------- Channel L2 norm ------------------------------
__global__ void k_l2norm(float* __restrict__ p, int C, int batchCh) {
  int i = blockIdx.x * 256 + threadIdx.x;       // 9216 pixels
  if (i >= BB * HWP) return;
  int b = i / HWP, pix = i % HWP;
  float* base = p + (size_t)b * batchCh * HWP + pix;
  float s = 0.f;
  for (int c = 0; c < C; ++c) { float v = base[(size_t)c * HWP]; s += v * v; }
  float inv = 1.f / fmaxf(sqrtf(s), 1e-12f);
  for (int c = 0; c < C; ++c) base[(size_t)c * HWP] *= inv;
}

// ------------------- Weight pre-pack: [O][I][K][K] -> f16 [t][O][I] --------
__global__ void k_wprep(const float* __restrict__ w, _Float16* __restrict__ o,
                        int Cout, int Cin, int KK, int total) {
  int id = blockIdx.x * 256 + threadIdx.x;
  if (id >= total) return;
  int t = id / (Cout * Cin);
  int rem = id - t * Cout * Cin;
  int oc = rem / Cin, ci = rem - oc * Cin;
  o[id] = (_Float16)w[((size_t)oc * Cin + ci) * KK + t];
}

// --------------- Implicit-GEMM conv (1x1 / 3x3) via WMMA f16 --------------
// Block 128 = 4 waves; block tile = 64 oc x 32 flattened pixels.
// Each wave: 16 oc x 32 px (2 WMMA accumulators). The 64x32-half A tile is
// staged by the Tensor Data Mover (2D tile, row stride Cin) when available.
// grid = (B*H*W/32 = 288, Cout/64). act: 0 none, 1 silu.
__global__ void k_conv_wmma(const float* __restrict__ in, const _Float16* __restrict__ wT,
                            const float* __restrict__ bias, const float* __restrict__ res,
                            float* __restrict__ out, int Cin, int Cout, int Ksz, int act) {
  const int p0 = blockIdx.x * 32;
  const int wave = threadIdx.x >> 5;
  const int lane = threadIdx.x & 31;
  const int oc0 = blockIdx.y * 64 + wave * 16;
  const int pad = Ksz >> 1;
  const int KK = Ksz * Ksz;
  const int M = lane & 15;
  const int kbase = (lane >> 4) * 8;

  __shared__ __align__(16) _Float16 ldsA[64][32];   // [oc][K], natural order
  __shared__ __align__(16) _Float16 ldsB[32][32];   // [px][K], natural order
  __shared__ int s_b[32], s_y[32], s_x[32];

  if (threadIdx.x < 32) {
    int p = p0 + threadIdx.x;
    s_x[threadIdx.x] = p % WW;
    int yx = p / WW;
    s_y[threadIdx.x] = yx % HH;
    s_b[threadIdx.x] = yx / HH;
  }
  __syncthreads();

  v8f acc0 = {}, acc1 = {};

  for (int t = 0; t < KK; ++t) {
    const int dy = t / Ksz - pad;
    const int dx = t % Ksz - pad;
    const _Float16* wtt = wT + ((size_t)t * Cout + blockIdx.y * 64) * Cin;
    for (int ci0 = 0; ci0 < Cin; ci0 += 32) {
#if USE_TDM
      // ---- stage A via TDM: 2D tile, 64 rows x 32 halfs, stride Cin ------
      if (wave == 0) {
        unsigned long long ga = (unsigned long long)(uintptr_t)(wtt + ci0);
        u32x4 g0;
        g0[0] = 1u;                                       // count=1 descriptor
        g0[1] = (unsigned)(uintptr_t)&ldsA[0][0];         // lds_addr (bytes)
        g0[2] = (unsigned)(ga & 0xFFFFFFFFu);             // global_addr[31:0]
        g0[3] = (unsigned)((ga >> 32) & 0x01FFFFFFu)      // global_addr[56:32]
                | 0x80000000u;                            // type=2 ("image")
        i32x8 g1;
        g1[0] = 0x00010000;                               // data_size=2B
        g1[1] = (int)((Cin & 0xFFFF) << 16);              // tensor_dim0 lo
        g1[2] = (int)(((unsigned)Cin >> 16) | (64u << 16)); // dim0 hi | dim1 lo
        g1[3] = (int)(32 << 16);                          // dim1 hi | tile_dim0=32
        g1[4] = 64;                                       // tile_dim1=64
        g1[5] = Cin;                                      // dim0_stride lo
        g1[6] = 0;
        g1[7] = 0;
        i32x4 z4 = {0, 0, 0, 0};
        i32x8 z8 = {0, 0, 0, 0, 0, 0, 0, 0};
        __builtin_amdgcn_tensor_load_to_lds(g0, g1, z4, z4, z8, 0);
      }
#else
      // ---- stage A: plain 16B chunk copies -------------------------------
      for (int e = threadIdx.x; e < 256; e += 128) {
        int row = e >> 2, g = e & 3;
        uint4 chunk = *(const uint4*)(wtt + (size_t)row * Cin + ci0 + g * 8);
        *(uint4*)&ldsA[row][g * 8] = chunk;
      }
#endif
      // ---- stage B: activations 32 px x 32 ci, f32 -> f16 ----------------
      for (int e = threadIdx.x; e < 1024; e += 128) {
        int ci = e >> 5, px = e & 31;
        int yy = s_y[px] + dy, xx = s_x[px] + dx;
        float v = 0.f;
        if (yy >= 0 && yy < HH && xx >= 0 && xx < WW)
          v = in[(((size_t)s_b[px] * Cin + ci0 + ci) * HH + yy) * WW + xx];
        ldsB[px][ci] = (_Float16)v;
      }
#if USE_TDM
      if (wave == 0) __builtin_amdgcn_s_wait_tensorcnt(0);
#endif
      __syncthreads();

      __builtin_prefetch(wtt + (size_t)M * Cin + ci0 + 32, 0, 1);

      // A fragment: two aligned 16B chunks from natural [row][K] layout
      const v8h* arow = (const v8h*)&ldsA[wave * 16 + M][0];
      v8h alo = arow[2 * (kbase >> 3)];
      v8h ahi = arow[2 * (kbase >> 3) + 1];
      v16h afr;
#pragma unroll
      for (int h = 0; h < 8; ++h) { afr[h] = alo[h]; afr[h + 8] = ahi[h]; }

      const v16h* bp0 = (const v16h*)&ldsB[lane & 15][0];
      const v16h* bp1 = (const v16h*)&ldsB[16 + (lane & 15)][0];
      v16h b0 = bp0[lane >> 4];
      v16h b1 = bp1[lane >> 4];
      acc0 = __builtin_amdgcn_wmma_f32_16x16x32_f16(false, afr, false, b0,
                                                    (short)0, acc0, false, false);
      acc1 = __builtin_amdgcn_wmma_f32_16x16x32_f16(false, afr, false, b1,
                                                    (short)0, acc1, false, false);
      __syncthreads();
    }
  }

  // epilogue: bias / residual / silu, scatter store
  const int rowoff = (lane >= 16) ? 8 : 0;
#pragma unroll
  for (int r = 0; r < 8; ++r) {
    int oc = oc0 + r + rowoff;
    float bv = bias ? bias[oc] : 0.f;
#pragma unroll
    for (int j = 0; j < 2; ++j) {
      int px = j * 16 + (lane & 15);
      float v = (j == 0 ? acc0[r] : acc1[r]) + bv;
      size_t oi = (((size_t)s_b[px] * Cout + oc) * HH + s_y[px]) * WW + s_x[px];
      if (res) v += res[oi];
      if (act == 1) v = v / (1.f + __expf(-v));   // silu
      out[oi] = v;
    }
  }
}

// --------------------------- rfft2 stage 1 (rows) -------------------------
__global__ void k_rfft_s1(const float* __restrict__ in, float* __restrict__ s1re,
                          float* __restrict__ s1im) {
  int idx = blockIdx.x * 256 + threadIdx.x;
  if (idx >= FREQN) return;
  int kx = idx % FREQW;
  int yrow = (idx / FREQW) % HH;
  int bc = idx / MTOK;
  const float* row = in + ((size_t)bc * HH + yrow) * WW;
  float re = 0.f, im = 0.f;
  float w = -PI2 * (float)kx / 48.f;
  for (int xcol = 0; xcol < WW; ++xcol) {
    float s, c;
    __sincosf(w * (float)xcol, &s, &c);
    re += row[xcol] * c;
    im += row[xcol] * s;
  }
  s1re[idx] = re; s1im[idx] = im;
}

// ------- rfft2 stage 2 (cols) + polar + f16 store in attention layout -----
// transp==0: [bh][m][32] (token-major, for Q/K fragments)
// transp==1: [bh][32][m] (channel-major, for V fragments)
__global__ void k_rfft_s2(const float* __restrict__ s1re, const float* __restrict__ s1im,
                          _Float16* __restrict__ amp, _Float16* __restrict__ phas,
                          int transp) {
  int idx = blockIdx.x * 256 + threadIdx.x;
  if (idx >= FREQN) return;
  int kx = idx % FREQW;
  int ky = (idx / FREQW) % HH;
  int bc = idx / MTOK;
  float re = 0.f, im = 0.f;
  float w = -PI2 * (float)ky / 48.f;
  for (int yrow = 0; yrow < HH; ++yrow) {
    float s, c;
    __sincosf(w * (float)yrow, &s, &c);
    size_t si = ((size_t)bc * HH + yrow) * FREQW + kx;
    float tr = s1re[si], ti = s1im[si];
    re += tr * c - ti * s;
    im += tr * s + ti * c;
  }
  re *= (1.f / 48.f); im *= (1.f / 48.f);     // ortho
  int b = bc >> 8, c = bc & 255;
  int bh = b * 8 + (c >> 5), cd = c & 31, m = ky * FREQW + kx;
  size_t o = transp ? (((size_t)bh * 32 + cd) * MTOK + m)
                    : (((size_t)bh * MTOK + m) * 32 + cd);
  amp[o]  = (_Float16)sqrtf(re * re + im * im);
  phas[o] = (_Float16)atan2f(im, re);
}

// ------------------ Flash attention (per head, WMMA) ----------------------
// Q,K: f16 [32 bh][1200 m][32 c]; V: f16 [32 bh][32 c][1200 m];
// O: f32 [32 bh][32 c][1200 m]. One wave per (bh, 16-token m tile).
__global__ void k_flash_attn(const _Float16* __restrict__ Q, const _Float16* __restrict__ K,
                             const _Float16* __restrict__ V, float* __restrict__ O) {
  const int bh = blockIdx.x;
  const int m0 = blockIdx.y * 16;
  const int lane = threadIdx.x;
  const float scale = 0.17677669529663687f;   // 1/sqrt(32)

  const _Float16* Qb = Q + (size_t)bh * MTOK * 32;
  const _Float16* Kb = K + (size_t)bh * MTOK * 32;
  const _Float16* Vb = V + (size_t)bh * 32 * MTOK;

  const int M = lane & 15;
  const int kbase = (lane >> 4) * 8;
  const int kb2 = (lane >= 16) ? 16 : 0;

  // Q tile as A fragment: row m0+M, two 8-half chunks (16B aligned loads)
  const v8h* qrow = (const v8h*)(Qb + (size_t)(m0 + M) * 32);
  v8h qlo = qrow[2 * (kbase >> 3)];
  v8h qhi = qrow[2 * (kbase >> 3) + 1];
  v16h aq;
#pragma unroll
  for (int h = 0; h < 8; ++h) { aq[h] = qlo[h]; aq[h + 8] = qhi[h]; }

  v8f oc0 = {}, oc1 = {};
  float rmax[8], rsum[8];
#pragma unroll
  for (int r = 0; r < 8; ++r) { rmax[r] = -1e30f; rsum[r] = 0.f; }

  __shared__ _Float16 pl[16][32];
  for (int e = lane; e < 256; e += 32) pl[e >> 4][16 + (e & 15)] = (_Float16)0.f;
  __syncthreads();

  for (int n0 = 0; n0 < MTOK; n0 += 16) {
    // K tile as B fragment: one contiguous 32B vector per lane
    v16h bk = ((const v16h*)(Kb + (size_t)(n0 + (lane & 15)) * 32))[lane >> 4];
    v8f sim = {};
    sim = __builtin_amdgcn_wmma_f32_16x16x32_f16(false, aq, false, bk,
                                                 (short)0, sim, false, false);
    // online softmax across the 16-lane half
    float corr[8];
#pragma unroll
    for (int r = 0; r < 8; ++r) {
      float v = sim[r] * scale;
      sim[r] = v;
      float mx = v;
      mx = fmaxf(mx, __shfl_xor(mx, 1));
      mx = fmaxf(mx, __shfl_xor(mx, 2));
      mx = fmaxf(mx, __shfl_xor(mx, 4));
      mx = fmaxf(mx, __shfl_xor(mx, 8));
      float nm = fmaxf(rmax[r], mx);
      corr[r] = __expf(rmax[r] - nm);
      rmax[r] = nm;
      float p = __expf(sim[r] - nm);
      sim[r] = p;
      float ts = p;
      ts += __shfl_xor(ts, 1);
      ts += __shfl_xor(ts, 2);
      ts += __shfl_xor(ts, 4);
      ts += __shfl_xor(ts, 8);
      rsum[r] = rsum[r] * corr[r] + ts;
      oc0[r] *= corr[r];
      oc1[r] *= corr[r];
    }
    __syncthreads();
    const int rowoff = (lane >= 16) ? 8 : 0;
#pragma unroll
    for (int r = 0; r < 8; ++r) pl[r + rowoff][lane & 15] = (_Float16)sim[r];
    __syncthreads();

    // P as A fragment (upper K half zero-padded)
    const v8h* prow = (const v8h*)&pl[lane & 15][0];
    v8h plo = prow[2 * (kbase >> 3)];
    v8h phi = prow[2 * (kbase >> 3) + 1];
    v16h pa;
#pragma unroll
    for (int h = 0; h < 8; ++h) { pa[h] = plo[h]; pa[h + 8] = phi[h]; }

    // V^T tiles as B fragments: contiguous 32B per lane (channel-major V)
    v16h bv0 = {}, bv1 = {};
    if (n0 + kb2 < MTOK) {
      bv0 = *(const v16h*)(Vb + (size_t)(0  + (lane & 15)) * MTOK + n0 + kb2);
      bv1 = *(const v16h*)(Vb + (size_t)(16 + (lane & 15)) * MTOK + n0 + kb2);
    }
    oc0 = __builtin_amdgcn_wmma_f32_16x16x32_f16(false, pa, false, bv0,
                                                 (short)0, oc0, false, false);
    oc1 = __builtin_amdgcn_wmma_f32_16x16x32_f16(false, pa, false, bv1,
                                                 (short)0, oc1, false, false);
    __syncthreads();
  }

  const int rowoff = (lane >= 16) ? 8 : 0;
  float* Ob = O + (size_t)bh * 32 * MTOK;
#pragma unroll
  for (int r = 0; r < 8; ++r) {
    float inv = 1.f / rsum[r];
    int m = m0 + r + rowoff;
    Ob[(size_t)(0  + (lane & 15)) * MTOK + m] = oc0[r] * inv;
    Ob[(size_t)(16 + (lane & 15)) * MTOK + m] = oc1[r] * inv;
  }
}

// ---------------- inverse FFT stage 1: polar -> rows (over kx) ------------
__global__ void k_ifft_s1(const float* __restrict__ amp, const float* __restrict__ phas,
                          float* __restrict__ Tre, float* __restrict__ Tim) {
  int idx = blockIdx.x * 256 + threadIdx.x;
  if (idx >= SPX) return;
  int xcol = idx % WW;
  int ky = (idx / WW) % HH;
  int bc = idx / HWP;
  int b = bc >> 8, c = bc & 255;
  size_t hb = (((size_t)(b * 8 + (c >> 5)) * 32) + (c & 31)) * MTOK;
  float re = 0.f, im = 0.f;
  float w = PI2 * (float)xcol / 48.f;
  for (int kx = 0; kx < WW; ++kx) {
    float xr, xi;
    if (kx < FREQW) {
      size_t o = hb + ky * FREQW + kx;
      float a = amp[o], p = phas[o];
      float sp, cp; __sincosf(p, &sp, &cp);
      xr = a * cp; xi = a * sp;
    } else {
      int ky2 = (48 - ky) % 48;
      int kx2 = 48 - kx;
      size_t o = hb + ky2 * FREQW + kx2;
      float a = amp[o], p = phas[o];
      float sp, cp; __sincosf(p, &sp, &cp);
      xr = a * cp; xi = -a * sp;              // conjugate
    }
    float s, cc;
    __sincosf(w * (float)kx, &s, &cc);
    re += xr * cc - xi * s;
    im += xr * s + xi * cc;
  }
  Tre[idx] = re; Tim[idx] = im;
}

// ---------------- inverse FFT stage 2: cols (over ky), real out -----------
__global__ void k_ifft_s2(const float* __restrict__ Tre, const float* __restrict__ Tim,
                          float* __restrict__ out) {
  int idx = blockIdx.x * 256 + threadIdx.x;
  if (idx >= SPX) return;
  int xcol = idx % WW;
  int yrow = (idx / WW) % HH;
  int bc = idx / HWP;
  float re = 0.f;
  float w = PI2 * (float)yrow / 48.f;
  for (int ky = 0; ky < HH; ++ky) {
    float s, c;
    __sincosf(w * (float)ky, &s, &c);
    size_t ti = ((size_t)bc * HH + ky) * WW + xcol;
    re += Tre[ti] * c - Tim[ti] * s;
  }
  out[idx] = re * (1.f / 48.f);               // ortho
}

// ---------------------------------------------------------------------------
extern "C" void kernel_launch(void* const* d_in, const int* in_sizes, int n_in,
                              void* d_out, int out_size, void* d_ws, size_t ws_size,
                              hipStream_t stream) {
  const float* x      = (const float*)d_in[0];
  const float* cond   = (const float*)d_in[1];
  const float* gn_w   = (const float*)d_in[2];
  const float* gn_b   = (const float*)d_in[3];
  const float* q_dw   = (const float*)d_in[4];
  const float* q_pw_w = (const float*)d_in[5];
  const float* q_pw_b = (const float*)d_in[6];
  const float* kv_dw  = (const float*)d_in[7];
  const float* kv_pw_w= (const float*)d_in[8];
  const float* kv_pw_b= (const float*)d_in[9];
  const float* ao_w   = (const float*)d_in[10];
  const float* ao_b   = (const float*)d_in[11];
  const float* w1     = (const float*)d_in[12];
  const float* w2     = (const float*)d_in[13];
  const float* w3     = (const float*)d_in[14];
  const float* b3     = (const float*)d_in[15];
  float* out = (float*)d_out;
  float* ws  = (float*)d_ws;

  // ---------------- workspace layout (float offsets) ----------------------
  float* xn    = ws;                               // SPX
  float* t0    = ws + (size_t)SPX;                 // SPX
  float* q     = ws + (size_t)2 * SPX;             // SPX
  float* kv    = ws + (size_t)3 * SPX;             // 2*SPX
  float* a0    = ws + (size_t)5 * SPX;             // SPX
  float* ampO  = ws + (size_t)6 * SPX;             // FREQN
  float* phasO = ampO + (size_t)FREQN;             // FREQN
  float* s1re  = phasO + (size_t)FREQN;            // FREQN
  float* s1im  = s1re + (size_t)FREQN;             // FREQN
  // f16 pool
  _Float16* hp = (_Float16*)(s1im + (size_t)FREQN);
  _Float16* ampQh  = hp;
  _Float16* phasQh = hp + (size_t)FREQN;
  _Float16* ampKh  = hp + (size_t)2 * FREQN;
  _Float16* phasKh = hp + (size_t)3 * FREQN;
  _Float16* ampVh  = hp + (size_t)4 * FREQN;
  _Float16* phasVh = hp + (size_t)5 * FREQN;
  _Float16* wPool  = hp + (size_t)6 * FREQN;
  _Float16* wt_qpw = wPool;                        // 65536
  _Float16* wt_kv  = wPool + 65536;                // 131072
  _Float16* wt_ao  = wPool + 196608;               // 65536
  _Float16* wt_w1  = wPool + 262144;               // 1179648
  _Float16* wt_w2  = wPool + 1441792;              // 1179648
  _Float16* wt_w3  = wPool + 2621440;              // 65536
  // reuse (dead after attention): iFFT temps and FFN activations
  float* Tre = q;                                  // SPX
  float* Tim = kv;                                 // SPX
  float* h1  = kv;                                 // 2*SPX
  float* h2  = q;                                  // SPX

  const dim3 blk256(256), blk128(128), blk32(32);
  const int gSpx = (SPX + 255) / 256;              // 9216
  const int gFrq = (FREQN + 255) / 256;            // 4800
  const dim3 gConv(288, 4), gConvWide(288, 8);
  const dim3 gAttn(32, 75);

  // 0) pack all conv weights to f16 [t][Cout][Cin]
  k_wprep<<<(65536 + 255) / 256, blk256, 0, stream>>>(q_pw_w, wt_qpw, 256, 256, 1, 65536);
  k_wprep<<<(131072 + 255) / 256, blk256, 0, stream>>>(kv_pw_w, wt_kv, 512, 256, 1, 131072);
  k_wprep<<<(65536 + 255) / 256, blk256, 0, stream>>>(ao_w, wt_ao, 256, 256, 1, 65536);
  k_wprep<<<(1179648 + 255) / 256, blk256, 0, stream>>>(w1, wt_w1, 512, 256, 9, 1179648);
  k_wprep<<<(1179648 + 255) / 256, blk256, 0, stream>>>(w2, wt_w2, 256, 512, 9, 1179648);
  k_wprep<<<(65536 + 255) / 256, blk256, 0, stream>>>(w3, wt_w3, 256, 256, 1, 65536);

  // 1) groupnorm
  k_groupnorm<<<128, blk256, 0, stream>>>(x, gn_w, gn_b, xn);
  // 2) q branch
  k_dw3x3<<<gSpx, blk256, 0, stream>>>(xn, q_dw, t0);
  k_conv_wmma<<<gConv, blk128, 0, stream>>>(t0, wt_qpw, q_pw_b, nullptr, q, 256, 256, 1, 0);
  // 3) kv branch
  k_dw3x3<<<gSpx, blk256, 0, stream>>>(cond, kv_dw, t0);
  k_conv_wmma<<<gConvWide, blk128, 0, stream>>>(t0, wt_kv, kv_pw_b, nullptr, kv, 256, 512, 1, 0);
  // 4) channel L2 norms
  k_l2norm<<<36, blk256, 0, stream>>>(q, 256, 256);
  k_l2norm<<<36, blk256, 0, stream>>>(kv, 256, 512);
  k_l2norm<<<36, blk256, 0, stream>>>(kv + (size_t)256 * HWP, 256, 512);
  // 5) rfft2 -> f16 amp/phase in attention layouts (Q/K token-major, V ch-major)
  k_rfft_s1<<<gFrq, blk256, 0, stream>>>(q, s1re, s1im);
  k_rfft_s2<<<gFrq, blk256, 0, stream>>>(s1re, s1im, ampQh, phasQh, 0);
  k_rfft_s1<<<gFrq, blk256, 0, stream>>>(kv, s1re, s1im);
  k_rfft_s2<<<gFrq, blk256, 0, stream>>>(s1re, s1im, ampKh, phasKh, 0);
  k_rfft_s1<<<gFrq, blk256, 0, stream>>>(kv + (size_t)256 * HWP, s1re, s1im);
  k_rfft_s2<<<gFrq, blk256, 0, stream>>>(s1re, s1im, ampVh, phasVh, 1);
  // 6) two WMMA flash attentions (amplitude, phase)
  k_flash_attn<<<gAttn, blk32, 0, stream>>>(ampQh, ampKh, ampVh, ampO);
  k_flash_attn<<<gAttn, blk32, 0, stream>>>(phasQh, phasKh, phasVh, phasO);
  // 7) polar + irfft2 -> spatial
  k_ifft_s1<<<gSpx, blk256, 0, stream>>>(ampO, phasO, Tre, Tim);
  k_ifft_s2<<<gSpx, blk256, 0, stream>>>(Tre, Tim, t0);
  // 8) attn_out = conv1x1(out) + bias + xn
  k_conv_wmma<<<gConv, blk128, 0, stream>>>(t0, wt_ao, ao_b, xn, a0, 256, 256, 1, 0);
  // 9) FFN
  k_conv_wmma<<<gConvWide, blk128, 0, stream>>>(a0, wt_w1, nullptr, nullptr, h1, 256, 512, 3, 1);
  k_conv_wmma<<<gConv, blk128, 0, stream>>>(h1, wt_w2, nullptr, nullptr, h2, 512, 256, 3, 0);
  k_conv_wmma<<<gConv, blk128, 0, stream>>>(h2, wt_w3, b3, a0, out, 256, 256, 1, 0);
}
